// GlobalSpatialAttention_84980222918707
// MI455X (gfx1250) — compile-verified, matrix operand order checked by reading, not run on hardware
//
#include <hip/hip_runtime.h>
#include <hip/hip_bf16.h>
#include <math.h>

// ---------------------------------------------------------------------------
// Global spatial attention, fused flash-style for gfx1250 (MI455X, wave32).
//   Prep   : pack projection weights (dw folded, f32->f16) into WMMA
//            B-operand lane layout (one 32B v16h load per lane thereafter).
//   Phase 0: QKV projection (WMMA f16), writes q,k f16 row-major [N,HW,64]
//            and v transposed [N,64,HW] f16.
//   Phase 1: softmax row stats, per-lane online max/sum, software-pipelined
//            tile loads; stores fused stat l[i]=m[i]+log z[i].
//   Phase 2: recompute S tiles (K tiles pipelined one iteration ahead),
//            P=exp(S-l) in-register, PV accumulate (WMMA), att [N,64,HW] f32.
//   Phase 3: output projection + residual gating (WMMA, pipelined B tiles).
// ---------------------------------------------------------------------------

typedef __attribute__((ext_vector_type(16))) _Float16 v16h;
typedef __attribute__((ext_vector_type(8)))  _Float16 v8h;
typedef __attribute__((ext_vector_type(8)))  float    v8f;

#define NB   4
#define HW   4096
#define CIN  256
#define CR   64

// WMMA D=A*B+C, A 16x32 f16 (lane m = row m), B 32x16 f16 (lane n = col n),
// C/D 16x16 f32.
__device__ __forceinline__ v8f wmma16(v16h a, v16h b, v8f c) {
  return __builtin_amdgcn_wmma_f32_16x16x32_f16(
      /*neg_a=*/false, a, /*neg_b=*/false, b,
      /*c_mod=*/(short)0, c, /*reuse_a=*/false, /*reuse_b=*/false);
}

// Load 16 f16 elements of one row per the 16-bit 16x32 operand layout:
// halves[0..7] = row[kbase+half8+0..7], halves[8..15] = row[kbase+16+half8+..].
__device__ __forceinline__ v16h load_row16(const _Float16* __restrict__ rowp,
                                           int kbase, int half8) {
  const v8h lo = *reinterpret_cast<const v8h*>(rowp + kbase + half8);
  const v8h hi = *reinterpret_cast<const v8h*>(rowp + kbase + 16 + half8);
  v16h r;
#pragma unroll
  for (int t = 0; t < 8; ++t) { r[t] = lo[t]; r[8 + t] = hi[t]; }
  return r;
}

// Branchless per-lane online softmax update over 8 row-slots.
__device__ __forceinline__ void online_update(v8f s, float* m, float* z) {
#pragma unroll
  for (int vv = 0; vv < 8; ++vv) {
    const float val = s[vv];
    const float mn  = fmaxf(m[vv], val);
    z[vv] = z[vv] * __expf(m[vv] - mn) + __expf(val - mn);
    m[vv] = mn;
  }
}

// ---------------------------------------------------------------------------
// Prep: pack weights into WMMA B-operand layout.
//   wqkv: 96 tiles  (o in {q,k,v}) x (jt 0..3) x (ks 0..7), 512 halves each.
//   wa  : 32 tiles  (jt 0..15) x (kh 0..1),                512 halves each.
// grid = 128 blocks, 32 threads.
// ---------------------------------------------------------------------------
__global__ __launch_bounds__(32) void pack_weights_kernel(
    const float* __restrict__ dw_q, const float* __restrict__ pw_q,
    const float* __restrict__ dw_k, const float* __restrict__ pw_k,
    const float* __restrict__ dw_v, const float* __restrict__ pw_v,
    const float* __restrict__ pw_a,
    _Float16* __restrict__ wqkv, _Float16* __restrict__ wa) {
  const int lane  = threadIdx.x & 31;
  const int col16 = lane & 15;
  const int half8 = (lane >> 4) * 8;
  const int b = blockIdx.x;
  if (b < 96) {  // qkv tile: b = o*32 + jt*8 + ks
    const int o = b >> 5, rem = b & 31, jt = rem >> 3, ks = rem & 7;
    const float* dw = (o == 0) ? dw_q : (o == 1) ? dw_k : dw_v;
    const float* pw = (o == 0) ? pw_q : (o == 1) ? pw_k : pw_v;
    const int col = jt * 16 + col16;
    const int kb  = ks * 32;
    _Float16* dst = wqkv + (size_t)b * 512 + lane * 16;
#pragma unroll
    for (int t = 0; t < 8; ++t) {
      const int c0 = kb + half8 + t;
      const int c1 = kb + 16 + half8 + t;
      dst[t]     = (_Float16)(dw[c0] * pw[c0 * CR + col]);
      dst[8 + t] = (_Float16)(dw[c1] * pw[c1 * CR + col]);
    }
  } else {       // pw_a tile: r = jt*2 + kh
    const int r = b - 96;
    const int jt = r >> 1, kh2 = r & 1;
    const int col = jt * 16 + col16;
    const int cb  = kh2 * 32;
    _Float16* dst = wa + (size_t)r * 512 + lane * 16;
#pragma unroll
    for (int t = 0; t < 8; ++t) {
      dst[t]     = (_Float16)pw_a[(cb + half8 + t) * CIN + col];
      dst[8 + t] = (_Float16)pw_a[(cb + 16 + half8 + t) * CIN + col];
    }
  }
}

// ---------------------------------------------------------------------------
// Phase 0: q/k/v projections.  grid = NB*HW/16 blocks, 32 threads (1 wave).
// All 8 packed-B tiles of a chain preloaded before the serial WMMA chain.
// ---------------------------------------------------------------------------
__global__ __launch_bounds__(32) void qkv_proj_kernel(
    const float* __restrict__ x,
    const _Float16* __restrict__ wqkv,
    const float* __restrict__ b_q, const float* __restrict__ b_k,
    const float* __restrict__ b_v,
    _Float16* __restrict__ qh, _Float16* __restrict__ kh,
    _Float16* __restrict__ vth) {
  const int lane  = threadIdx.x & 31;
  const int row   = lane & 15;
  const int half8 = (lane >> 4) * 8;
  const long r0   = (long)blockIdx.x * 16;

  // A tiles: 16 rows of x, K=256 split into 8 steps of 32.
  const float* xrow = x + (r0 + row) * CIN;
  v16h A[8];
#pragma unroll
  for (int ks = 0; ks < 8; ++ks) {
    const int kb = ks * 32;
#pragma unroll
    for (int t = 0; t < 8; ++t) {
      A[ks][t]     = (_Float16)xrow[kb + half8 + t];
      A[ks][8 + t] = (_Float16)xrow[kb + 16 + half8 + t];
    }
  }

  const float* bss[3] = {b_q, b_k, b_v};
  const int rowsel = 8 * (lane >> 4);

  for (int o = 0; o < 3; ++o) {
#pragma unroll
    for (int jt = 0; jt < 4; ++jt) {
      const int col = jt * 16 + (lane & 15);
      v16h B[8];
#pragma unroll
      for (int ks = 0; ks < 8; ++ks)
        B[ks] = *reinterpret_cast<const v16h*>(
            wqkv + ((size_t)(o * 32 + jt * 8 + ks)) * 512 + lane * 16);
      v8f acc = {};
#pragma unroll
      for (int ks = 0; ks < 8; ++ks) acc = wmma16(A[ks], B[ks], acc);
      const float bias = bss[o][col];
      if (o < 2) {
        _Float16* dst = (o == 0) ? qh : kh;
#pragma unroll
        for (int vv = 0; vv < 8; ++vv) {
          const long rr = r0 + rowsel + vv;
          dst[rr * CR + col] = (_Float16)(acc[vv] + bias);
        }
      } else {  // v stored transposed: vT[n, c, pos]
#pragma unroll
        for (int vv = 0; vv < 8; ++vv) {
          const long rr  = r0 + rowsel + vv;
          const long n   = rr >> 12;
          const long pos = rr & (HW - 1);
          vth[(n * CR + col) * (long)HW + pos] = (_Float16)(acc[vv] + bias);
        }
      }
    }
  }
}

// ---------------------------------------------------------------------------
// Phase 1: softmax row stats.  grid = NB * HW/16, 32 threads.
// Software-pipelined: iteration jt computes on tiles loaded at jt-1.
// Stores l[i] = m[i] + log z[i].
// ---------------------------------------------------------------------------
__global__ __launch_bounds__(32) void rowstats_kernel(
    const _Float16* __restrict__ qh, const _Float16* __restrict__ kh,
    float* __restrict__ rowl) {
  const int blk   = blockIdx.x;
  const int n     = blk >> 8;
  const int ib    = blk & 255;
  const int lane  = threadIdx.x & 31;
  const int row   = lane & 15;
  const int half8 = (lane >> 4) * 8;

  const _Float16* krow = kh + ((long)n * HW + ib * 16 + row) * CR;
  const v16h A0 = load_row16(krow, 0, half8);
  const v16h A1 = load_row16(krow, 32, half8);

  float m[8], z[8];
#pragma unroll
  for (int vv = 0; vv < 8; ++vv) { m[vv] = -INFINITY; z[vv] = 0.0f; }

  const _Float16* qn = qh + (long)n * HW * CR;
  v16h B0 = load_row16(qn + row * CR, 0, half8);
  v16h B1 = load_row16(qn + row * CR, 32, half8);
  for (int jt = 0; jt < HW / 16 - 1; ++jt) {
    const _Float16* qnext = qn + ((long)(jt + 1) * 16 + row) * CR;
    const v16h nB0 = load_row16(qnext, 0, half8);   // issued before use of B0
    const v16h nB1 = load_row16(qnext, 32, half8);
    v8f s = {};
    s = wmma16(A0, B0, s);
    s = wmma16(A1, B1, s);
    online_update(s, m, z);
    B0 = nB0;
    B1 = nB1;
  }
  {  // peeled last tile
    v8f s = {};
    s = wmma16(A0, B0, s);
    s = wmma16(A1, B1, s);
    online_update(s, m, z);
  }
  // Cross-lane merge (16-lane groups share the same rows).
#pragma unroll
  for (int vv = 0; vv < 8; ++vv) {
    float mg = m[vv];
    mg = fmaxf(mg, __shfl_xor(mg, 1, 32));
    mg = fmaxf(mg, __shfl_xor(mg, 2, 32));
    mg = fmaxf(mg, __shfl_xor(mg, 4, 32));
    mg = fmaxf(mg, __shfl_xor(mg, 8, 32));
    float zz = z[vv] * __expf(m[vv] - mg);
    zz += __shfl_xor(zz, 1, 32);
    zz += __shfl_xor(zz, 2, 32);
    zz += __shfl_xor(zz, 4, 32);
    zz += __shfl_xor(zz, 8, 32);
    m[vv] = mg + __logf(zz);  // fused stat l = m + log z
  }
  if ((lane & 15) == 0) {  // lanes 0 and 16 own rows 0-7 / 8-15
    const int rb = ib * 16 + 8 * (lane >> 4);
#pragma unroll
    for (int vv = 0; vv < 8; ++vv) rowl[n * HW + rb + vv] = m[vv];
  }
}

// ---------------------------------------------------------------------------
// Phase 2: att[c,j] = sum_i vT[c,i] * exp(S[i,j]-l[i]).
// grid = NB * HW/16 (j-blocks), 32 threads.  K tiles pipelined one iteration
// ahead; Vt tiles + stats issued before the S/exp work they overlap with.
// P tile feeds the PV WMMA directly (D layout == B layout, col match).
// ---------------------------------------------------------------------------
__global__ __launch_bounds__(32) void attn_pv_kernel(
    const _Float16* __restrict__ qh, const _Float16* __restrict__ kh,
    const _Float16* __restrict__ vth,
    const float* __restrict__ rowl, float* __restrict__ att) {
  const int blk    = blockIdx.x;
  const int n      = blk >> 8;
  const int jb     = blk & 255;
  const int lane   = threadIdx.x & 31;
  const int row    = lane & 15;
  const int half8  = (lane >> 4) * 8;
  const int rowsel = 8 * (lane >> 4);

  const _Float16* qrow = qh + ((long)n * HW + jb * 16 + row) * CR;
  const v16h Bq0 = load_row16(qrow, 0, half8);
  const v16h Bq1 = load_row16(qrow, 32, half8);

  const _Float16* kn  = kh + (long)n * HW * CR;
  const _Float16* vtn = vth + (long)n * CR * HW;
  const float*    ln  = rowl + n * HW;

  v8f accv[4] = {{}, {}, {}, {}};

  // Preload K operand tiles for it = 0.
  v16h Ak0a = load_row16(kn + row * CR, 0, half8);
  v16h Ak0b = load_row16(kn + row * CR, 32, half8);
  v16h Ak1a = load_row16(kn + (16 + row) * CR, 0, half8);
  v16h Ak1b = load_row16(kn + (16 + row) * CR, 32, half8);

  for (int it = 0; it < HW / 32; ++it) {
    const int i0    = it * 32;
    const int inext = (it < HW / 32 - 1) ? i0 + 32 : i0;
    __builtin_prefetch(kn + ((long)inext + 32 + row) * CR, 0, 3);

    // Independent loads issued up front: fused stats + Vt tiles.
    const v8f lv0 = *reinterpret_cast<const v8f*>(ln + i0 + rowsel);
    const v8f lv1 = *reinterpret_cast<const v8f*>(ln + i0 + 16 + rowsel);
    v16h Av[4];
#pragma unroll
    for (int cb = 0; cb < 4; ++cb)
      Av[cb] = load_row16(vtn + ((long)cb * 16 + row) * HW + i0, 0, half8);

    // Next iteration's K tiles (consumed next loop trip).
    const _Float16* kr0 = kn + ((long)inext + row) * CR;
    const _Float16* kr1 = kn + ((long)inext + 16 + row) * CR;
    const v16h nAk0a = load_row16(kr0, 0, half8);
    const v16h nAk0b = load_row16(kr0, 32, half8);
    const v16h nAk1a = load_row16(kr1, 0, half8);
    const v16h nAk1b = load_row16(kr1, 32, half8);

    // S tiles (loads from a previous trip) -> P (exp co-executes with WMMA).
    v8f s0 = {};
    s0 = wmma16(Ak0a, Bq0, s0);
    s0 = wmma16(Ak0b, Bq1, s0);
    v8f s1 = {};
    s1 = wmma16(Ak1a, Bq0, s1);
    s1 = wmma16(Ak1b, Bq1, s1);
    v16h P;  // B-operand layout (K=32 i-values x 16 j-cols), in-lane mapping
#pragma unroll
    for (int vv = 0; vv < 8; ++vv) {
      P[vv]     = (_Float16)__expf(s0[vv] - lv0[vv]);
      P[8 + vv] = (_Float16)__expf(s1[vv] - lv1[vv]);
    }
#pragma unroll
    for (int cb = 0; cb < 4; ++cb) accv[cb] = wmma16(Av[cb], P, accv[cb]);

    Ak0a = nAk0a; Ak0b = nAk0b; Ak1a = nAk1a; Ak1b = nAk1b;
  }

  const int col = jb * 16 + (lane & 15);
#pragma unroll
  for (int cb = 0; cb < 4; ++cb)
#pragma unroll
    for (int vv = 0; vv < 8; ++vv)
      att[((long)n * CR + cb * 16 + rowsel + vv) * HW + col] = accv[cb][vv];
}

// ---------------------------------------------------------------------------
// Phase 3: out = g * ((att4*dw_a) @ pw_a + b_a) + g.
// Raw reshape [N,Cr,HW]->[N,H,W,Cr]: att4 row (p=h*64+w) is the contiguous
// 64-float slice att[n, h, w*64 : w*64+64].  grid = NB*HW/16, 32 threads.
// B tiles pipelined one jt ahead.
// ---------------------------------------------------------------------------
__global__ __launch_bounds__(32) void out_proj_kernel(
    const float* __restrict__ att,
    const float* __restrict__ dw_a, const _Float16* __restrict__ wa,
    const float* __restrict__ b_a,
    const float* __restrict__ g, float* __restrict__ out) {
  const int lane   = threadIdx.x & 31;
  const int half8  = (lane >> 4) * 8;
  const int rowsel = 8 * (lane >> 4);
  const long r0 = (long)blockIdx.x * 16;
  const long rr = r0 + (lane & 15);
  const long n  = rr >> 12;
  const int  p  = (int)(rr & (HW - 1));
  const int  h  = p >> 6, w = p & 63;

  const float* arow = att + ((n * CR + h) * (long)HW) + (long)w * 64;
  v16h A0, A1;
#pragma unroll
  for (int t = 0; t < 8; ++t) {
    A0[t]     = (_Float16)(arow[half8 + t]      * dw_a[half8 + t]);
    A0[8 + t] = (_Float16)(arow[16 + half8 + t] * dw_a[16 + half8 + t]);
    A1[t]     = (_Float16)(arow[32 + half8 + t] * dw_a[32 + half8 + t]);
    A1[8 + t] = (_Float16)(arow[48 + half8 + t] * dw_a[48 + half8 + t]);
  }

  v16h B0 = *reinterpret_cast<const v16h*>(wa + (size_t)0 * 512 + lane * 16);
  v16h B1 = *reinterpret_cast<const v16h*>(wa + (size_t)1 * 512 + lane * 16);
#pragma unroll
  for (int jt = 0; jt < CIN / 16; ++jt) {
    const int jn = (jt < CIN / 16 - 1) ? jt + 1 : jt;
    const v16h nB0 = *reinterpret_cast<const v16h*>(
        wa + (size_t)(jn * 2 + 0) * 512 + lane * 16);
    const v16h nB1 = *reinterpret_cast<const v16h*>(
        wa + (size_t)(jn * 2 + 1) * 512 + lane * 16);
    const int col = jt * 16 + (lane & 15);
    v8f acc = {};
    acc = wmma16(A0, B0, acc);
    acc = wmma16(A1, B1, acc);
    const float bias = b_a[col];
#pragma unroll
    for (int vv = 0; vv < 8; ++vv) {
      const long orow = r0 + rowsel + vv;
      const long idx  = orow * CIN + col;
      const float gg  = g[idx];
      out[idx] = gg * (acc[vv] + bias) + gg;
    }
    B0 = nB0;
    B1 = nB1;
  }
}

// ---------------------------------------------------------------------------
extern "C" void kernel_launch(void* const* d_in, const int* in_sizes, int n_in,
                              void* d_out, int out_size, void* d_ws, size_t ws_size,
                              hipStream_t stream) {
  const float* x    = (const float*)d_in[0];   // feature_maps [4,64,64,256]
  const float* g    = (const float*)d_in[1];   // global_channel_output
  const float* dw_q = (const float*)d_in[2];
  const float* pw_q = (const float*)d_in[3];
  const float* b_q  = (const float*)d_in[4];
  const float* dw_k = (const float*)d_in[5];
  const float* pw_k = (const float*)d_in[6];
  const float* b_k  = (const float*)d_in[7];
  const float* dw_v = (const float*)d_in[8];
  const float* pw_v = (const float*)d_in[9];
  const float* b_v  = (const float*)d_in[10];
  const float* dw_a = (const float*)d_in[11];
  const float* pw_a = (const float*)d_in[12];
  const float* b_a  = (const float*)d_in[13];
  float* out = (float*)d_out;

  // Workspace layout (~10.3 MB):
  //   q,k  f16 [N,HW,64]   : 2 MB each
  //   vT   f16 [N,64,HW]   : 2 MB
  //   wqkv f16 packed tiles: 96 KB
  //   wa   f16 packed tiles: 32 KB
  //   rowl f32 [N,HW]      : 64 KB   (32B aligned for v8f loads)
  //   att  f32 [N,64,HW]   : 4 MB
  char* w = (char*)d_ws;
  const size_t qkvBytes = (size_t)NB * HW * CR * sizeof(_Float16);
  _Float16* qh   = (_Float16*)(w);
  _Float16* kh   = (_Float16*)(w + qkvBytes);
  _Float16* vth  = (_Float16*)(w + 2 * qkvBytes);
  _Float16* wqkv = (_Float16*)(w + 3 * qkvBytes);
  _Float16* wa   = (_Float16*)(w + 3 * qkvBytes + 96 * 512 * sizeof(_Float16));
  float*    rowl = (float*)(w + 3 * qkvBytes + 128 * 512 * sizeof(_Float16));
  float*    att  = rowl + (size_t)NB * HW;

  const dim3 blk(32);
  const int nBlocks = NB * HW / 16;  // 1024

  pack_weights_kernel<<<128, blk, 0, stream>>>(
      dw_q, pw_q, dw_k, pw_k, dw_v, pw_v, pw_a, wqkv, wa);
  qkv_proj_kernel<<<nBlocks, blk, 0, stream>>>(
      x, wqkv, b_q, b_k, b_v, qh, kh, vth);
  rowstats_kernel<<<nBlocks, blk, 0, stream>>>(qh, kh, rowl);
  attn_pv_kernel<<<nBlocks, blk, 0, stream>>>(qh, kh, vth, rowl, att);
  out_proj_kernel<<<nBlocks, blk, 0, stream>>>(att, dw_a, wa, b_a, g, out);
}
